// LSTMRegression_79697413145097
// MI455X (gfx1250) — matrix operand order repaired and checked by reading, not compiled, not run on hardware
//
#include <hip/hip_runtime.h>

#define T_LEN 2048
#define B_SZ  512
#define C_IN  8
#define H_SZ  20
#define G4H   80

typedef __attribute__((ext_vector_type(16))) __bf16 v16bf;
typedef __attribute__((ext_vector_type(8)))  float  v8f;

union Frag {
    v16bf v;
    unsigned int u[8];
    unsigned short s[16];
};
union UV4 {
    uint4 v;
    unsigned int a[4];
};

__device__ __forceinline__ unsigned short f2bf(float f) {
    unsigned int u = __float_as_uint(f);
    u += 0x7FFFu + ((u >> 16) & 1u);          // round-to-nearest-even
    return (unsigned short)(u >> 16);
}
__device__ __forceinline__ float sigm(float x) {
    return __builtin_amdgcn_rcpf(1.0f + __expf(-x));
}
__device__ __forceinline__ float tanh_(float x) {
    // tanh(x) = 1 - 2/(1+e^{2x})
    return fmaf(-2.0f, __builtin_amdgcn_rcpf(1.0f + __expf(2.0f * x)), 1.0f);
}

// One wave per 16-batch tile; persistent over all 2048 timesteps, both layers fused.
__global__ __launch_bounds__(32) void lstm2_fused_kernel(
    const float* __restrict__ x,
    const float* __restrict__ Wih0, const float* __restrict__ Whh0,
    const float* __restrict__ bih0, const float* __restrict__ bhh0,
    const float* __restrict__ Wih1, const float* __restrict__ Whh1,
    const float* __restrict__ bih1, const float* __restrict__ bhh1,
    float* __restrict__ out)
{
    const int lane = threadIdx.x;          // 0..31 (wave32)
    const int b0   = blockIdx.x * 16;      // batch tile base
    const int nn   = lane & 15;            // column / row-in-half index
    const bool hiL = lane >= 16;

    __shared__ __align__(16) float          Gbuf[16 * G4H];   // gate tile, interleaved (i,f,g,o) per hidden
    __shared__ __align__(16) unsigned short H1[16 * 24];      // h state, bf16, 24-col padded rows (48B, 16B-aligned slices)
    __shared__ __align__(16) unsigned short H2[16 * 24];

    for (int i = lane; i < 16 * 24; i += 32) { H1[i] = 0; H2[i] = 0; }
    __syncthreads();

    // ---- Pack weights into B-matrix (32x16 bf16) fragments, gate-interleaved columns ----
    // column c = 4*hidden + gate  ->  original weight row R = gate*20 + hidden
    // B layout: lanes 0-15 hold K=0..7,16..23; lanes 16-31 hold K=8..15,24..31 (e<8 -> K=8h+e, e>=8 -> K=16+8h+e-8).
    Frag Wc[5], Wi1[5], Wh1[5];
    float bias1[5], bias2[5];
    const int kb16 = hiL ? 16 : 0;
#pragma unroll
    for (int j = 0; j < 5; ++j) {
        int c    = j * 16 + nn;
        int hid  = c >> 2;
        int gate = c & 3;
        int R    = gate * H_SZ + hid;
        bias1[j] = bih0[R] + bhh0[R];
        bias2[j] = bih1[R] + bhh1[R];
#pragma unroll
        for (int e = 0; e < 16; ++e) {
            int k = (e < 8 ? 0 : 8) + (hiL ? 8 : 0) + e;
            // layer-1 concatenated K: [x (0..7) | h1 (8..27) | zero (28..31)]
            float w0 = (k < 8) ? Wih0[R * C_IN + k]
                     : ((k < 28) ? Whh0[R * H_SZ + (k - 8)] : 0.0f);
            Wc[j].s[e] = f2bf(w0);
            float w1 = (k < H_SZ) ? Wih1[R * H_SZ + k] : 0.0f;   // h1 padded to K=32
            Wi1[j].s[e] = f2bf(w1);
            float w2 = (k < H_SZ) ? Whh1[R * H_SZ + k] : 0.0f;   // h2 padded to K=32
            Wh1[j].s[e] = f2bf(w2);
        }
    }

    // Cell state: 320 (m,n) pairs spread over 32 lanes, 10 each.
    float c1[10], c2[10];
#pragma unroll
    for (int q = 0; q < 10; ++q) { c1[q] = 0.0f; c2[q] = 0.0f; }

    // Prefetch x_t=0 for row m = nn (lanes >=16 load same rows; harmless, keeps EXEC full)
    const float* xr0 = x + ((size_t)0 * B_SZ + b0 + nn) * C_IN;
    float4 xa = ((const float4*)xr0)[0];
    float4 xb = ((const float4*)xr0)[1];

    const int mrow = hiL ? 8 : 0;   // C/D: vgpr r -> row r (lanes 0-15) or 8+r (lanes 16-31)
    const int m    = nn;            // A-fragment row owned by this lane

    for (int t = 0; t < T_LEN; ++t) {
        // x_t -> 8 bf16 packed in 4 dwords (A elements e0..7 for lanes 0-15)
        unsigned int xp0 = (unsigned)f2bf(xa.x) | ((unsigned)f2bf(xa.y) << 16);
        unsigned int xp1 = (unsigned)f2bf(xa.z) | ((unsigned)f2bf(xa.w) << 16);
        unsigned int xp2 = (unsigned)f2bf(xb.x) | ((unsigned)f2bf(xb.y) << 16);
        unsigned int xp3 = (unsigned)f2bf(xb.z) | ((unsigned)f2bf(xb.w) << 16);

        // prefetch x_{t+1} (overlaps with WMMA + update latency)
        int tn = (t + 1 < T_LEN) ? (t + 1) : t;
        const float* xr = x + ((size_t)tn * B_SZ + b0 + nn) * C_IN;
        float4 nxa = ((const float4*)xr)[0];
        float4 nxb = ((const float4*)xr)[1];

        // ---- layer-1 A fragment: [x | h1 | 0], K=32, ISA 16-bit A layout ----
        UV4 ha, hb, hc;
        ha.v = *(const uint4*)(H1 + m * 24);        // h1 cols 0..7
        hb.v = *(const uint4*)(H1 + m * 24 + 8);    // h1 cols 8..15
        hc.v = *(const uint4*)(H1 + m * 24 + 16);   // h1 cols 16..23 (20..23 = 0 pad)
        Frag A1;
        A1.u[0] = hiL ? ha.a[0] : xp0;   // lo half: e0..7 -> K0..7 (=x) or K8..15 (=h1 0..7)
        A1.u[1] = hiL ? ha.a[1] : xp1;
        A1.u[2] = hiL ? ha.a[2] : xp2;
        A1.u[3] = hiL ? ha.a[3] : xp3;
#pragma unroll
        for (int i = 0; i < 4; ++i)      // hi half: e8..15 -> K16..23 (=h1 8..15) or K24..31 (=h1 16..23)
            A1.u[4 + i] = hiL ? hc.a[i] : hb.a[i];

        // ---- layer-1 gates: one WMMA per 16-col tile ----
        v8f g1[5];
#pragma unroll
        for (int j = 0; j < 5; ++j) {
            v8f cc;
#pragma unroll
            for (int r = 0; r < 8; ++r) cc[r] = bias1[j];
            g1[j] = __builtin_amdgcn_wmma_f32_16x16x32_bf16(
                        false, A1.v, false, Wc[j].v, (short)0, cc, false, false);
        }

        // spill gate tiles to LDS in plain [16][80] layout
#pragma unroll
        for (int j = 0; j < 5; ++j)
#pragma unroll
            for (int r = 0; r < 8; ++r)
                Gbuf[(mrow + r) * G4H + j * 16 + nn] = g1[j][r];
        __syncthreads();

        // ---- layer-1 pointwise update: (i,f,g,o) contiguous -> one b128 per unit ----
#pragma unroll
        for (int q = 0; q < 10; ++q) {
            int p  = q * 32 + lane;
            int pm = p / H_SZ, pn = p % H_SZ;
            float4 gv = *(const float4*)(Gbuf + pm * G4H + 4 * pn);
            float ig = sigm(gv.x), fg = sigm(gv.y), gg = tanh_(gv.z), og = sigm(gv.w);
            c1[q] = fg * c1[q] + ig * gg;
            float h = og * tanh_(c1[q]);
            H1[pm * 24 + pn] = f2bf(h);
        }
        __syncthreads();

        // ---- layer-2 A fragments: h1 and h2, each zero-padded to K=32 ----
        UV4 p1a, p1b, p1c, p2a, p2b, p2c;
        p1a.v = *(const uint4*)(H1 + m * 24);
        p1b.v = *(const uint4*)(H1 + m * 24 + 8);
        p1c.v = *(const uint4*)(H1 + m * 24 + 16);
        p2a.v = *(const uint4*)(H2 + m * 24);
        p2b.v = *(const uint4*)(H2 + m * 24 + 8);
        p2c.v = *(const uint4*)(H2 + m * 24 + 16);
        Frag Ah1, Ah2;
#pragma unroll
        for (int i = 0; i < 4; ++i) {
            Ah1.u[i]     = hiL ? p1b.a[i] : p1a.a[i];  // e0..7: K0..7 or K8..15
            Ah1.u[4 + i] = hiL ? 0u       : p1c.a[i];  // e8..15: K16..23 or K24..31(=0)
            Ah2.u[i]     = hiL ? p2b.a[i] : p2a.a[i];
            Ah2.u[4 + i] = hiL ? 0u       : p2c.a[i];
        }

        // ---- layer-2 gates: two chained WMMAs per tile ----
        v8f g2[5];
#pragma unroll
        for (int j = 0; j < 5; ++j) {
            v8f cc;
#pragma unroll
            for (int r = 0; r < 8; ++r) cc[r] = bias2[j];
            cc = __builtin_amdgcn_wmma_f32_16x16x32_bf16(
                     false, Ah2.v, false, Wh1[j].v, (short)0, cc, false, false);
            g2[j] = __builtin_amdgcn_wmma_f32_16x16x32_bf16(
                     false, Ah1.v, false, Wi1[j].v, (short)0, cc, false, false);
        }

#pragma unroll
        for (int j = 0; j < 5; ++j)
#pragma unroll
            for (int r = 0; r < 8; ++r)
                Gbuf[(mrow + r) * G4H + j * 16 + nn] = g2[j][r];
        __syncthreads();

        // ---- layer-2 pointwise update + output store ----
#pragma unroll
        for (int q = 0; q < 10; ++q) {
            int p  = q * 32 + lane;
            int pm = p / H_SZ, pn = p % H_SZ;
            float4 gv = *(const float4*)(Gbuf + pm * G4H + 4 * pn);
            float ig = sigm(gv.x), fg = sigm(gv.y), gg = tanh_(gv.z), og = sigm(gv.w);
            c2[q] = fg * c2[q] + ig * gg;
            float h = og * tanh_(c2[q]);
            H2[pm * 24 + pn] = f2bf(h);
            out[((size_t)t * B_SZ + b0 + pm) * H_SZ + pn] = h;
        }
        __syncthreads();

        xa = nxa; xb = nxb;
    }
}

extern "C" void kernel_launch(void* const* d_in, const int* in_sizes, int n_in,
                              void* d_out, int out_size, void* d_ws, size_t ws_size,
                              hipStream_t stream) {
    (void)in_sizes; (void)n_in; (void)out_size; (void)d_ws; (void)ws_size;
    const float* x    = (const float*)d_in[0];
    const float* Wih0 = (const float*)d_in[1];
    const float* Whh0 = (const float*)d_in[2];
    const float* bih0 = (const float*)d_in[3];
    const float* bhh0 = (const float*)d_in[4];
    const float* Wih1 = (const float*)d_in[5];
    const float* Whh1 = (const float*)d_in[6];
    const float* bih1 = (const float*)d_in[7];
    const float* bhh1 = (const float*)d_in[8];
    float* out = (float*)d_out;

    dim3 grid(B_SZ / 16);   // 32 independent batch tiles
    dim3 block(32);         // one wave32 per tile
    hipLaunchKernelGGL(lstm2_fused_kernel, grid, block, 0, stream,
                       x, Wih0, Whh0, bih0, bhh0, Wih1, Whh1, bih1, bhh1, out);
}
